// GraphAttentionLayer_36455682408739
// MI455X (gfx1250) — matrix operand order repaired and checked by reading, not compiled, not run on hardware
//
#include <hip/hip_runtime.h>

// MI455X / gfx1250, wave32. Fused GAT layer:
//   k_wt  : W -> W^T f16 (WMMA-B friendly)
//   k_wh  : Wh = h@W via v_wmma_f32_16x16x32_f16; si/sj reductions; Wh^T f16 out
//   k_attn: flash-style online softmax over adj + P@Wh via WMMA, ELU epilogue
//
// Workspace layout (bytes):
//   [0,        4 MiB)  Wh^T  f16   [B][F][N]
//   [4 MiB,   +64 KiB) si    f32   [B][N]
//   [.. ,     +64 KiB) sj    f32   [B][N]
//   [.. ,     +32 KiB) W^T   f16   [F_out][F_in]
// total ~4.16 MiB

typedef __attribute__((ext_vector_type(16))) _Float16 v16h;
typedef __attribute__((ext_vector_type(8)))  float    v8f;

#define NB 4
#define NN 4096
#define NF 128
#define LRELU_ALPHA 0.2f
#define NEG_INF_VAL -1.0e9f

static __device__ __forceinline__ v8f wmma_f16(v16h a, v16h b, v8f c) {
  // D = A(16x32 f16) * B(32x16 f16) + C(16x16 f32)
  return __builtin_amdgcn_wmma_f32_16x16x32_f16(
      /*neg_a=*/false, a, /*neg_b=*/false, b,
      /*c_mod=*/(short)0, c, /*reuse_a=*/false, /*reuse_b=*/false);
}

// ---------------------------------------------------------------- k_wt
// W^T in f16: WT[n][k] = (f16) W[k][n]; rows contiguous in K for B-matrix loads.
__global__ void __launch_bounds__(256) k_wt(const float* __restrict__ W,
                                            _Float16* __restrict__ WT) {
  int idx = blockIdx.x * blockDim.x + threadIdx.x;  // 0..16383
  int n = idx >> 7, k = idx & 127;
  WT[n * NF + k] = (_Float16)W[k * NF + n];
}

// ---------------------------------------------------------------- k_wh
// One wave per 16-row tile. 1024 tiles total -> 256 blocks x 4 waves.
__global__ void __launch_bounds__(128) k_wh(
    const float* __restrict__ h, const _Float16* __restrict__ WT,
    const float* __restrict__ a,
    _Float16* __restrict__ WhT, float* __restrict__ si, float* __restrict__ sj) {
  const int lane   = threadIdx.x & 31;
  const int wave   = blockIdx.x * 4 + (threadIdx.x >> 5);
  const int b      = wave >> 8;          // 256 tiles per batch
  const int i_base = (wave & 255) << 4;
  const int n  = lane & 15;
  const int hi = lane >> 4;

  const float* hrow = h + ((size_t)b * NN + i_base + n) * NF;

  v8f acc[8];
  const v8f zero = {0.f, 0.f, 0.f, 0.f, 0.f, 0.f, 0.f, 0.f};
#pragma unroll
  for (int nt = 0; nt < 8; ++nt) acc[nt] = zero;

#pragma unroll
  for (int c = 0; c < 4; ++c) {  // K = 128 in chunks of 32
    // A-matrix layout: lane<16 row=n holds K = {c*32+[0,8), c*32+[16,24)},
    //                  lane>=16 same row holds K = {+8..+16, +24..+32}
    const float4* hp = (const float4*)(hrow + c * 32);
    float4 f0 = hp[2 * hi], f1 = hp[2 * hi + 1];
    float4 f2 = hp[4 + 2 * hi], f3 = hp[5 + 2 * hi];
    float tf[16] = {f0.x, f0.y, f0.z, f0.w, f1.x, f1.y, f1.z, f1.w,
                    f2.x, f2.y, f2.z, f2.w, f3.x, f3.y, f3.z, f3.w};
    v16h A;
#pragma unroll
    for (int t = 0; t < 16; ++t) A[t] = (_Float16)tf[t];

#pragma unroll
    for (int nt = 0; nt < 8; ++nt) {
      // B-matrix: lane holds column nt*16+n, K = c*32 + {8*hi+[0,8), 16+8*hi+[0,8)}
      typedef __attribute__((ext_vector_type(8))) _Float16 v8h_;
      const _Float16* bq = WT + (nt * 16 + n) * NF + c * 32 + hi * 8;
      v8h_ b0 = *(const v8h_*)bq, b1 = *(const v8h_*)(bq + 16);
      v16h Bm = __builtin_shufflevector(b0, b1, 0,1,2,3,4,5,6,7,8,9,10,11,12,13,14,15);
      acc[nt] = wmma_f16(A, Bm, acc[nt]);
    }
  }

  float ai[8], aj[8];
#pragma unroll
  for (int nt = 0; nt < 8; ++nt) {
    ai[nt] = a[nt * 16 + n];
    aj[nt] = a[NF + nt * 16 + n];
  }

#pragma unroll
  for (int v = 0; v < 8; ++v) {
    const int row = i_base + v + 8 * hi;  // C-layout: VGPR v -> M = v + 8*hi
    float ps = 0.f, pj = 0.f;
#pragma unroll
    for (int nt = 0; nt < 8; ++nt) {
      float w = acc[nt][v];
      ps += w * ai[nt];
      pj += w * aj[nt];
      // Wh^T f16 for kernel-3 B-matrix loads (feature-major)
      WhT[((size_t)(b * NF + nt * 16 + n)) * NN + row] = (_Float16)w;
    }
    // reduce across the 16 lanes of this half (masks 1..8 stay in-half)
#pragma unroll
    for (int msk = 1; msk < 16; msk <<= 1) {
      ps += __shfl_xor(ps, msk, 32);
      pj += __shfl_xor(pj, msk, 32);
    }
    if (n == 0) {
      si[b * NN + row] = ps;
      sj[b * NN + row] = pj;
    }
  }
}

// ---------------------------------------------------------------- k_attn
// One wave per 16-row tile; online softmax over j in chunks of 32 (WMMA K),
// P tile fed directly into v_wmma_f32_16x16x32_f16 against resident Wh^T.
__global__ void __launch_bounds__(128) k_attn(
    const int* __restrict__ adj, const float* __restrict__ si,
    const float* __restrict__ sj, const _Float16* __restrict__ WhT,
    float* __restrict__ out) {
  const int lane   = threadIdx.x & 31;
  const int wave   = blockIdx.x * 4 + (threadIdx.x >> 5);
  const int b      = wave >> 8;
  const int i_base = (wave & 255) << 4;
  const int n  = lane & 15;   // stats/P row for this lane = n
  const int hi = lane >> 4;

  const float s_i     = si[b * NN + i_base + n];
  const int*   adjrow = adj + ((size_t)b * NN + i_base + n) * NN;
  const float* sjb    = sj + b * NN;
  const _Float16* wbase = WhT + (size_t)b * NF * NN + hi * 8;

  v8f acc[8];
  const v8f zero = {0.f, 0.f, 0.f, 0.f, 0.f, 0.f, 0.f, 0.f};
#pragma unroll
  for (int nt = 0; nt < 8; ++nt) acc[nt] = zero;
  float mrun = -1.0e30f, lrun = 0.f;

  for (int jb = 0; jb < NN; jb += 32) {
    // This lane's 16 j-offsets: [8*hi, 8*hi+8) and [16+8*hi, 24+8*hi)
    const int4* ap = (const int4*)(adjrow + jb);
    int4 a0 = ap[2 * hi], a1 = ap[2 * hi + 1];
    int4 a2 = ap[4 + 2 * hi], a3 = ap[5 + 2 * hi];
    const float4* sp = (const float4*)(sjb + jb);
    float4 s0 = sp[2 * hi], s1 = sp[2 * hi + 1];
    float4 s2 = sp[4 + 2 * hi], s3 = sp[5 + 2 * hi];

    if (jb + 32 < NN) {  // global_prefetch_b8 for next adj chunk
      __builtin_prefetch(adjrow + jb + 32 + 8 * hi, 0, 1);
      __builtin_prefetch(adjrow + jb + 48 + 8 * hi, 0, 1);
    }

    int av[16] = {a0.x, a0.y, a0.z, a0.w, a1.x, a1.y, a1.z, a1.w,
                  a2.x, a2.y, a2.z, a2.w, a3.x, a3.y, a3.z, a3.w};
    float sv[16] = {s0.x, s0.y, s0.z, s0.w, s1.x, s1.y, s1.z, s1.w,
                    s2.x, s2.y, s2.z, s2.w, s3.x, s3.y, s3.z, s3.w};

    float x[16];
    float cm = -1.0e30f;
#pragma unroll
    for (int t = 0; t < 16; ++t) {
      float e = s_i + sv[t];
      e = (e > 0.f) ? e : LRELU_ALPHA * e;   // LeakyReLU
      e = (av[t] > 0) ? e : NEG_INF_VAL;     // adjacency mask
      x[t] = e;
      cm = fmaxf(cm, e);
    }
    cm = fmaxf(cm, __shfl_xor(cm, 16, 32));  // combine both half-rows
    const float mnew = fmaxf(mrun, cm);

    float p[16];
    float psum = 0.f;
#pragma unroll
    for (int t = 0; t < 16; ++t) {
      p[t] = __expf(x[t] - mnew);
      psum += p[t];
    }
    psum += __shfl_xor(psum, 16, 32);

    if (__any((int)(cm > mrun))) {  // wave-uniform rescale; skipped once max stabilizes
      const float scale = __expf(mrun - mnew);
      lrun *= scale;
      float sc[8];
#pragma unroll
      for (int v = 0; v < 8; ++v) sc[v] = __shfl(scale, v + 8 * hi, 32);
#pragma unroll
      for (int nt = 0; nt < 8; ++nt)
#pragma unroll
        for (int v = 0; v < 8; ++v) acc[nt][v] *= sc[v];
    }
    lrun += psum;
    mrun = mnew;

    v16h P;  // A-matrix tile: row n, K per the 16-bit A layout
#pragma unroll
    for (int t = 0; t < 16; ++t) P[t] = (_Float16)p[t];

    const _Float16* wp = wbase + jb;
#pragma unroll
    for (int nt = 0; nt < 8; ++nt) {
      // B-matrix: column nt*16+n, K = {8*hi+[0,8), 16+8*hi+[0,8)} -> b128 pair
      typedef __attribute__((ext_vector_type(8))) _Float16 v8h_;
      const _Float16* bq = wp + (size_t)(nt * 16 + n) * NN;
      v8h_ b0 = *(const v8h_*)bq, b1 = *(const v8h_*)(bq + 16);
      v16h Bm = __builtin_shufflevector(b0, b1, 0,1,2,3,4,5,6,7,8,9,10,11,12,13,14,15);
      acc[nt] = wmma_f16(P, Bm, acc[nt]);
    }
  }

  float rl[8];
#pragma unroll
  for (int v = 0; v < 8; ++v) rl[v] = 1.f / __shfl(lrun, v + 8 * hi, 32);

#pragma unroll
  for (int v = 0; v < 8; ++v) {
    const int row = i_base + v + 8 * hi;
    float* orow = out + ((size_t)b * NN + row) * NF;
#pragma unroll
    for (int nt = 0; nt < 8; ++nt) {
      float y = acc[nt][v] * rl[v];
      y = (y > 0.f) ? y : expm1f(y);  // ELU
      orow[nt * 16 + n] = y;
    }
  }
}

// ---------------------------------------------------------------- launch
extern "C" void kernel_launch(void* const* d_in, const int* in_sizes, int n_in,
                              void* d_out, int out_size, void* d_ws, size_t ws_size,
                              hipStream_t stream) {
  const float* h   = (const float*)d_in[0];  // [4,4096,128] f32
  const int*   adj = (const int*)d_in[1];    // [4,4096,4096] i32
  const float* W   = (const float*)d_in[2];  // [128,128] f32
  const float* a   = (const float*)d_in[3];  // [256] f32
  float* out = (float*)d_out;                // [4,4096,128] f32

  char* ws = (char*)d_ws;
  _Float16* WhT = (_Float16*)ws;                                   // 4 MiB
  float*    si  = (float*)(ws + (size_t)4 * 1024 * 1024);          // 64 KiB
  float*    sj  = (float*)(ws + (size_t)4 * 1024 * 1024 + 65536);  // 64 KiB
  _Float16* WT  = (_Float16*)(ws + (size_t)4 * 1024 * 1024 + 131072);  // 32 KiB

  k_wt<<<64, 256, 0, stream>>>(W, WT);
  k_wh<<<256, 128, 0, stream>>>(h, WT, a, WhT, si, sj);
  k_attn<<<256, 128, 0, stream>>>(adj, si, sj, WhT, out);
}
